// LightLTCSeizNet_41506563949040
// MI455X (gfx1250) — compile-verified, matrix operand order checked by reading, not run on hardware
//
#include <hip/hip_runtime.h>
#include <math.h>

typedef __attribute__((ext_vector_type(16))) _Float16 v16h;
typedef __attribute__((ext_vector_type(8)))  float    v8f;

#define B_    64
#define C_    22
#define T_    32768
#define CF_   176          // C*8
#define TL_   1025
#define L_    64
#define KENC_ 1232         // 176*7
#define KENCP_ 1248        // padded to mult of 32
#define H1_   64
#define H2_   48
#define DA_   32
#define RR_   (B_*TL_)     // 65600 rows (exact multiple of 16)
#define RP_   (RR_+16)     // padded rows for Q/K (attention halo reads)
#define DT_   0.04f

// ---- WMMA wave32 fragment K-index (16-bit A/B, 16x16x32) ----
__device__ __forceinline__ int wk(int j, int half) {
  int p = j >> 1;
  return ((p >> 2) << 4) + half * 8 + ((p & 3) << 1) + (j & 1);
}
__device__ __forceinline__ float gelu_exact(float x) {
  return 0.5f * x * (1.0f + erff(x * 0.70710678118654752f));
}
__device__ __forceinline__ float sigmoidf_(float x) { return 1.0f / (1.0f + expf(-x)); }

// ============ init: zero Q/K pad rows ============
__global__ void init_kernel(_Float16* Qb, _Float16* Kb) {
  int idx = blockIdx.x * blockDim.x + threadIdx.x;
  if (idx < 16 * DA_) {
    Qb[(size_t)RR_ * DA_ + idx] = (_Float16)0.f;
    Kb[(size_t)RR_ * DA_ + idx] = (_Float16)0.f;
  }
}

// ============ channel energy: mean |x| over T ============
__global__ void energy_kernel(const float* __restrict__ x, float* __restrict__ chE) {
  int c = blockIdx.x, b = blockIdx.y;
  const float* xp = x + ((size_t)b * C_ + c) * T_;
  float s = 0.f;
  for (int i = threadIdx.x; i < T_; i += 256) s += fabsf(xp[i]);
  __shared__ float red[256];
  red[threadIdx.x] = s; __syncthreads();
  for (int off = 128; off > 0; off >>= 1) {
    if (threadIdx.x < off) red[threadIdx.x] += red[threadIdx.x + off];
    __syncthreads();
  }
  if (threadIdx.x == 0) chE[b * C_ + c] = red[0] / (float)T_;
}

// ============ gate MLP: sigmoid(gelu(e@W1^T+b1)@W2^T+b2) ============
__global__ void gate_kernel(const float* __restrict__ chE,
                            const float* __restrict__ w1, const float* __restrict__ b1,
                            const float* __restrict__ w2, const float* __restrict__ b2,
                            float* __restrict__ gate) {
  int b = blockIdx.x; int j = threadIdx.x; // 32 threads
  __shared__ float e[C_]; __shared__ float h[11];
  if (j < C_) e[j] = chE[b * C_ + j];
  __syncthreads();
  if (j < 11) {
    float a = b1[j];
    for (int i = 0; i < C_; i++) a += e[i] * w1[j * C_ + i];
    h[j] = gelu_exact(a);
  }
  __syncthreads();
  if (j < C_) {
    float a = b2[j];
    for (int i = 0; i < 11; i++) a += h[i] * w2[j * 11 + i];
    gate[b * C_ + j] = sigmoidf_(a);
  }
}

// ============ weight prep to f16 (padded) ============
__global__ void prep_encw_kernel(const float* __restrict__ w, _Float16* __restrict__ out) {
  for (int idx = blockIdx.x * blockDim.x + threadIdx.x; idx < L_ * KENCP_;
       idx += gridDim.x * blockDim.x) {
    int o = idx / KENCP_, k = idx % KENCP_;
    out[idx] = (k < KENC_) ? (_Float16)w[o * KENC_ + k] : (_Float16)0.f;
  }
}
__global__ void prep_pad16_kernel(const float* __restrict__ w, _Float16* __restrict__ out,
                                  int N, int K, int Kp) {
  for (int idx = blockIdx.x * blockDim.x + threadIdx.x; idx < N * Kp;
       idx += gridDim.x * blockDim.x) {
    int n = idx / Kp, k = idx % Kp;
    out[idx] = (k < K) ? (_Float16)w[n * K + k] : (_Float16)0.f;
  }
}

// ============ grouped filterbank conv + BN + GELU (VALU, LDS-tiled) ============
#define FBTILE 128
#define FBWIN  (127*32 + 128) // 4192
__global__ void fb_conv_kernel(const float* __restrict__ x, const float* __restrict__ gate,
                               const float* __restrict__ fbw, const float* __restrict__ fbb,
                               const float* __restrict__ g, const float* __restrict__ be,
                               const float* __restrict__ mm, const float* __restrict__ vv,
                               _Float16* __restrict__ fbact) {
  int c = blockIdx.x, b = blockIdx.y;
  int tid = threadIdx.x; // 256
  __shared__ float w[8][128];
  __shared__ float xt[FBWIN];
  for (int i = tid; i < 8 * 128; i += 256)
    w[i >> 7][i & 127] = fbw[(c * 8 + (i >> 7)) * 128 + (i & 127)];
  float gv = gate[b * C_ + c];
  const float* xb = x + ((size_t)b * C_ + c) * T_;
  for (int tile = 0; tile < 9; tile++) {
    int base = tile * FBTILE * 32 - 64; // pad=64
    __syncthreads();
    for (int i = tid; i < FBWIN; i += 256) {
      int gi = base + i;
      xt[i] = (gi >= 0 && gi < T_) ? xb[gi] * gv : 0.f;
    }
    __syncthreads();
    for (int it = tid; it < 8 * FBTILE; it += 256) {
      int f = it >> 7, tl = it & 127;
      int t = tile * FBTILE + tl;
      if (t < TL_) {
        float acc = 0.f;
        const float* xr = xt + tl * 32;
        #pragma unroll 16
        for (int k = 0; k < 128; k++) acc += xr[k] * w[f][k];
        int o = c * 8 + f;
        float y = acc + fbb[o];
        y = (y - mm[o]) * (g[o] * rsqrtf(vv[o] + 1e-5f)) + be[o];
        fbact[((size_t)b * CF_ + o) * TL_ + t] = (_Float16)gelu_exact(y);
      }
    }
  }
}

// ============ encoder conv as WMMA GEMM (im2col on the fly) + BN + GELU ============
__global__ void enc_conv_wmma_kernel(const _Float16* __restrict__ fbact,
                                     const _Float16* __restrict__ encw,
                                     const float* __restrict__ eb,
                                     const float* __restrict__ g, const float* __restrict__ be,
                                     const float* __restrict__ mm, const float* __restrict__ vv,
                                     _Float16* __restrict__ enc16) {
  int nt = blockIdx.x;        // 65 t-tiles
  int mt = blockIdx.y;        // 4 o-tiles
  int b  = blockIdx.z;
  int lane = threadIdx.x;
  int half = lane >> 4, q = lane & 15;
  int t0 = nt * 16, m0 = mt * 16;
  v8f acc = {};
  for (int kc = 0; kc < KENCP_; kc += 32) {
    v16h a, bm;
    #pragma unroll
    for (int j = 0; j < 16; j++) {
      int K = kc + wk(j, half);
      a[j] = encw[(m0 + q) * KENCP_ + K];
      _Float16 v = (_Float16)0.f;
      if (K < KENC_) {
        int ci = K / 7, kk = K % 7;
        int t = t0 + q + kk - 3; // pad=3
        if (t >= 0 && t < TL_)
          v = fbact[((size_t)b * CF_ + ci) * TL_ + t];
      }
      bm[j] = v;
    }
    acc = __builtin_amdgcn_wmma_f32_16x16x32_f16(false, a, false, bm, (short)0, acc,
                                                 false, false);
  }
  #pragma unroll
  for (int v = 0; v < 8; v++) {
    int o = m0 + v + 8 * half;
    int t = t0 + q;
    if (t < TL_) {
      float y = acc[v] + eb[o];
      y = (y - mm[o]) * (g[o] * rsqrtf(vv[o] + 1e-5f)) + be[o];
      y = gelu_exact(y);
      enc16[((size_t)b * TL_ + t) * 64 + o] = (_Float16)y; // [B*TL][64] row-major
    }
  }
}

// ============ generic WMMA GEMM: out[r,n] = sum_k A[r,k]*W[n,k] + bias[n] ============
__global__ void gemm_xwT_kernel(const _Float16* __restrict__ A, int R, int Kd, int lda,
                                const _Float16* __restrict__ W, const float* __restrict__ bias,
                                float* __restrict__ outF, _Float16* __restrict__ outH, int ldc) {
  int rt = blockIdx.x, nt = blockIdx.y;
  int lane = threadIdx.x;
  int half = lane >> 4, q = lane & 15;
  int r0 = rt * 16, n0 = nt * 16;
  v8f acc = {};
  for (int kc = 0; kc < Kd; kc += 32) {
    v16h a, bm;
    #pragma unroll
    for (int j = 0; j < 16; j++) {
      int K = kc + wk(j, half);
      a[j]  = A[(size_t)(r0 + q) * lda + K];
      bm[j] = W[(n0 + q) * Kd + K];
    }
    acc = __builtin_amdgcn_wmma_f32_16x16x32_f16(false, a, false, bm, (short)0, acc,
                                                 false, false);
  }
  #pragma unroll
  for (int v = 0; v < 8; v++) {
    int r = r0 + v + 8 * half;
    int n = n0 + q;
    if (r < R) {
      float y = acc[v] + (bias ? bias[n] : 0.f);
      if (outF) outF[(size_t)r * ldc + n] = y;
      if (outH) outH[(size_t)r * ldc + n] = (_Float16)y;
    }
  }
}

// ============ LTC scan (sequential over TL, one workgroup per batch) ============
__global__ void ltc_scan_kernel(const float* __restrict__ xp, int ldx, int H,
                                const float* __restrict__ Wh, const float* __restrict__ mask,
                                const float* __restrict__ log_tau,
                                _Float16* __restrict__ hout, float* __restrict__ pooled) {
  int b = blockIdx.x;
  int j = threadIdx.x; // 64 threads
  __shared__ float W[H1_ * H1_];
  __shared__ float hs[H1_];
  for (int i = j; i < H * H; i += 64) W[i] = Wh[i] * mask[i];
  bool act = j < H;
  float alpha = 0.f, hj = 0.f, pool = 0.f;
  if (act) {
    float lt = log_tau[j];
    float sp = (lt > 20.f) ? lt : log1pf(expf(lt));
    float tau = fminf(fmaxf(sp + 1e-6f, 0.01f), 10.f);
    alpha = expf(-DT_ / tau);
  }
  hs[j] = 0.f;
  __syncthreads();
  const float* xb = xp + (size_t)b * TL_ * ldx;
  _Float16* hb = hout + (size_t)b * TL_ * 64;
  for (int t = 0; t < TL_; t++) {
    float pre = 0.f;
    if (act) {
      pre = xb[t * ldx + j];
      const float* wr = W + j * H;
      #pragma unroll 16
      for (int i = 0; i < H; i++) pre += hs[i] * wr[i];
    }
    __syncthreads();
    if (act) {
      hj = alpha * hj + (1.f - alpha) * tanhf(pre);
      hs[j] = hj;
      pool += hj;
    }
    hb[t * 64 + j] = act ? (_Float16)hj : (_Float16)0.f; // pad cols zeroed
    __syncthreads();
  }
  if (pooled && act) pooled[b * H + j] = pool / (float)TL_;
}

// ============ attention: WMMA scores + register flash softmax + column partials ====
__global__ void attn_kernel(const _Float16* __restrict__ Qb, const _Float16* __restrict__ Kb,
                            float* __restrict__ part) {
  int tt = blockIdx.x, b = blockIdx.y;
  int tid = threadIdx.x;                 // 256 = 8 waves
  int wave = tid >> 5, lane = tid & 31;
  int half = lane >> 4, q = lane & 15;
  int t0 = tt * 16;
  __shared__ _Float16 Qt[16][DA_];
  __shared__ float pmax[8][16], psum[8][16];
  __shared__ float gm[16], gs[16];
  for (int i = tid; i < 16 * DA_; i += 256) {
    int r = i >> 5, c2 = i & 31;
    Qt[r][c2] = Qb[((size_t)b * TL_ + (t0 + r)) * DA_ + c2]; // padded rows exist
  }
  __syncthreads();
  v16h a;
  #pragma unroll
  for (int j = 0; j < 16; j++) a[j] = Qt[q][wk(j, half)];
  const float scale = 0.17677669529663687f; // 1/sqrt(32)
  int cnt = (65 - wave + 7) / 8;           // s-tiles handled by this wave
  v8f scf[9];
  #pragma unroll
  for (int i = 0; i < 9; i++) {
    if (i < cnt) {
      int st = wave + 8 * i;
      v16h bm;
      #pragma unroll
      for (int j = 0; j < 16; j++)
        bm[j] = Kb[((size_t)b * TL_ + (st * 16 + q)) * DA_ + wk(j, half)];
      v8f acc = {};
      acc = __builtin_amdgcn_wmma_f32_16x16x32_f16(false, a, false, bm, (short)0, acc,
                                                   false, false);
      int s = st * 16 + q;
      #pragma unroll
      for (int v = 0; v < 8; v++) scf[i][v] = (s < TL_) ? acc[v] * scale : -INFINITY;
    } else {
      #pragma unroll
      for (int v = 0; v < 8; v++) scf[i][v] = -INFINITY;
    }
  }
  // stage 1: global row max
  float lmax[8];
  #pragma unroll
  for (int v = 0; v < 8; v++) {
    float m = -INFINITY;
    #pragma unroll
    for (int i = 0; i < 9; i++) m = fmaxf(m, scf[i][v]);
    for (int msk = 1; msk <= 8; msk <<= 1) m = fmaxf(m, __shfl_xor(m, msk, 32));
    lmax[v] = m;
  }
  if (q == 0) {
    #pragma unroll
    for (int v = 0; v < 8; v++) pmax[wave][v + 8 * half] = lmax[v];
  }
  __syncthreads();
  if (tid < 16) {
    float m = pmax[0][tid];
    for (int w2 = 1; w2 < 8; w2++) m = fmaxf(m, pmax[w2][tid]);
    gm[tid] = m;
  }
  __syncthreads();
  // stage 2: global row sum of exp
  #pragma unroll
  for (int v = 0; v < 8; v++) {
    float gmv = gm[v + 8 * half];
    float s2 = 0.f;
    #pragma unroll
    for (int i = 0; i < 9; i++) s2 += expf(scf[i][v] - gmv);
    for (int msk = 1; msk <= 8; msk <<= 1) s2 += __shfl_xor(s2, msk, 32);
    lmax[v] = s2; // reuse as partial sum
  }
  if (q == 0) {
    #pragma unroll
    for (int v = 0; v < 8; v++) psum[wave][v + 8 * half] = lmax[v];
  }
  __syncthreads();
  if (tid < 16) {
    float s2 = 0.f;
    for (int w2 = 0; w2 < 8; w2++) s2 += psum[w2][tid];
    gs[tid] = s2;
  }
  __syncthreads();
  // column accumulation: mean over valid rows t, write per-tile partials
  #pragma unroll
  for (int i = 0; i < 9; i++) {
    if (i < cnt) {
      float colsum = 0.f;
      #pragma unroll
      for (int v = 0; v < 8; v++) {
        int row = v + 8 * half;
        if (t0 + row < TL_) colsum += expf(scf[i][v] - gm[row]) / gs[row];
      }
      colsum += __shfl_xor(colsum, 16, 32); // combine halves (rows 0-7 + 8-15)
      int s = (wave + 8 * i) * 16 + q;
      if (half == 0 && s < TL_)
        part[((size_t)tt * B_ + b) * TL_ + s] = colsum * (1.0f / (float)TL_);
    }
  }
}

__global__ void att_reduce_kernel(const float* __restrict__ part, float* __restrict__ att) {
  int idx = blockIdx.x * blockDim.x + threadIdx.x;
  if (idx < B_ * TL_) {
    float s = 0.f;
    for (int tt = 0; tt < 65; tt++) s += part[(size_t)tt * (B_ * TL_) + idx];
    att[idx] = s;
  }
}

// ============ heads: pooled -> detect/pred ============
__global__ void heads_kernel(const float* __restrict__ pooled,
                             const float* __restrict__ dw1, const float* __restrict__ db1,
                             const float* __restrict__ dw2, const float* __restrict__ db2,
                             const float* __restrict__ pw1, const float* __restrict__ pb1,
                             const float* __restrict__ pw2, const float* __restrict__ pb2,
                             float* __restrict__ out) {
  int b = blockIdx.x, u = threadIdx.x; // 64 threads
  __shared__ float p[H2_];
  __shared__ float red[64];
  if (u < H2_) p[u] = pooled[b * H2_ + u];
  __syncthreads();
  float a = db1[u];
  for (int i = 0; i < H2_; i++) a += p[i] * dw1[u * H2_ + i];
  red[u] = gelu_exact(a) * dw2[u];
  __syncthreads();
  for (int off = 32; off > 0; off >>= 1) { if (u < off) red[u] += red[u + off]; __syncthreads(); }
  if (u == 0) out[b] = sigmoidf_(red[0] + db2[0]);
  __syncthreads();
  a = pb1[u];
  for (int i = 0; i < H2_; i++) a += p[i] * pw1[u * H2_ + i];
  red[u] = gelu_exact(a) * pw2[u];
  __syncthreads();
  for (int off = 32; off > 0; off >>= 1) { if (u < off) red[u] += red[u + off]; __syncthreads(); }
  if (u == 0) out[B_ + b] = sigmoidf_(red[0] + pb2[0]);
}

// ======================= host launcher =======================
extern "C" void kernel_launch(void* const* d_in, const int* in_sizes, int n_in,
                              void* d_out, int out_size, void* d_ws, size_t ws_size,
                              hipStream_t stream) {
  (void)in_sizes; (void)n_in; (void)out_size; (void)ws_size;
  const float* x        = (const float*)d_in[0];
  const float* gate_w1  = (const float*)d_in[1];
  const float* gate_b1  = (const float*)d_in[2];
  const float* gate_w2  = (const float*)d_in[3];
  const float* gate_b2  = (const float*)d_in[4];
  const float* fb_w     = (const float*)d_in[5];
  const float* fb_b     = (const float*)d_in[6];
  const float* bn_fb_g  = (const float*)d_in[7];
  const float* bn_fb_b  = (const float*)d_in[8];
  const float* bn_fb_m  = (const float*)d_in[9];
  const float* bn_fb_v  = (const float*)d_in[10];
  const float* enc_w    = (const float*)d_in[11];
  const float* enc_b    = (const float*)d_in[12];
  const float* bn_enc_g = (const float*)d_in[13];
  const float* bn_enc_b = (const float*)d_in[14];
  const float* bn_enc_m = (const float*)d_in[15];
  const float* bn_enc_v = (const float*)d_in[16];
  const float* Wx1      = (const float*)d_in[17];
  const float* Wh1      = (const float*)d_in[18];
  const float* b1v      = (const float*)d_in[19];
  const float* log_tau1 = (const float*)d_in[20];
  const float* Wx2      = (const float*)d_in[21];
  const float* Wh2      = (const float*)d_in[22];
  const float* b2v      = (const float*)d_in[23];
  const float* log_tau2 = (const float*)d_in[24];
  const float* q_w      = (const float*)d_in[25];
  const float* q_b      = (const float*)d_in[26];
  const float* k_w      = (const float*)d_in[27];
  const float* k_b      = (const float*)d_in[28];
  // d_in[29], d_in[30] = v_w, v_b: unused in the reference computation
  const float* det_w1   = (const float*)d_in[31];
  const float* det_b1   = (const float*)d_in[32];
  const float* det_w2   = (const float*)d_in[33];
  const float* det_b2   = (const float*)d_in[34];
  const float* pred_w1  = (const float*)d_in[35];
  const float* pred_b1  = (const float*)d_in[36];
  const float* pred_w2  = (const float*)d_in[37];
  const float* pred_b2  = (const float*)d_in[38];
  const float* mask1    = (const float*)d_in[39];
  const float* mask2    = (const float*)d_in[40];

  char* ws = (char*)d_ws;
  size_t off = 0;
  auto walloc = [&](size_t bytes) -> void* {
    void* p = ws + off;
    off = (off + bytes + 255) & ~(size_t)255;
    return p;
  };
  float*    chE    = (float*)   walloc(B_ * C_ * 4);
  float*    gateb  = (float*)   walloc(B_ * C_ * 4);
  _Float16* encw16 = (_Float16*)walloc((size_t)L_ * KENCP_ * 2);
  _Float16* wx116  = (_Float16*)walloc(H1_ * 64 * 2);
  _Float16* wx216  = (_Float16*)walloc(H2_ * 64 * 2);
  _Float16* qw16   = (_Float16*)walloc(DA_ * 64 * 2);
  _Float16* kw16   = (_Float16*)walloc(DA_ * 64 * 2);
  _Float16* fbact  = (_Float16*)walloc((size_t)B_ * CF_ * TL_ * 2);
  _Float16* enc16  = (_Float16*)walloc((size_t)RR_ * 64 * 2);
  float*    xp1    = (float*)   walloc((size_t)RR_ * 64 * 4);
  _Float16* h1_16  = (_Float16*)walloc((size_t)RR_ * 64 * 2);
  float*    xp2    = (float*)   walloc((size_t)RR_ * 48 * 4);
  _Float16* h2_16  = (_Float16*)walloc((size_t)RR_ * 64 * 2);
  _Float16* Qbuf   = (_Float16*)walloc((size_t)RP_ * DA_ * 2);
  _Float16* Kbuf   = (_Float16*)walloc((size_t)RP_ * DA_ * 2);
  float*    pooled = (float*)   walloc(B_ * H2_ * 4);
  float*    attp   = (float*)   walloc((size_t)65 * B_ * TL_ * 4);

  float* out   = (float*)d_out;
  float* att   = out + 2 * B_; // (detect[64], pred[64], att[64*1025])

  init_kernel<<<2, 256, 0, stream>>>(Qbuf, Kbuf);
  energy_kernel<<<dim3(C_, B_), 256, 0, stream>>>(x, chE);
  gate_kernel<<<B_, 32, 0, stream>>>(chE, gate_w1, gate_b1, gate_w2, gate_b2, gateb);
  prep_encw_kernel<<<128, 256, 0, stream>>>(enc_w, encw16);
  prep_pad16_kernel<<<32, 256, 0, stream>>>(Wx1, wx116, H1_, 64, 64);
  prep_pad16_kernel<<<32, 256, 0, stream>>>(Wx2, wx216, H2_, 64, 64);
  prep_pad16_kernel<<<16, 256, 0, stream>>>(q_w, qw16, DA_, H2_, 64);
  prep_pad16_kernel<<<16, 256, 0, stream>>>(k_w, kw16, DA_, H2_, 64);
  fb_conv_kernel<<<dim3(C_, B_), 256, 0, stream>>>(x, gateb, fb_w, fb_b, bn_fb_g, bn_fb_b,
                                                   bn_fb_m, bn_fb_v, fbact);
  enc_conv_wmma_kernel<<<dim3(65, 4, B_), 32, 0, stream>>>(fbact, encw16, enc_b, bn_enc_g,
                                                           bn_enc_b, bn_enc_m, bn_enc_v, enc16);
  gemm_xwT_kernel<<<dim3(RR_ / 16, 4), 32, 0, stream>>>(enc16, RR_, 64, 64, wx116, b1v,
                                                        xp1, nullptr, 64);
  ltc_scan_kernel<<<B_, 64, 0, stream>>>(xp1, 64, H1_, Wh1, mask1, log_tau1, h1_16, nullptr);
  gemm_xwT_kernel<<<dim3(RR_ / 16, 3), 32, 0, stream>>>(h1_16, RR_, 64, 64, wx216, b2v,
                                                        xp2, nullptr, 48);
  ltc_scan_kernel<<<B_, 64, 0, stream>>>(xp2, 48, H2_, Wh2, mask2, log_tau2, h2_16, pooled);
  gemm_xwT_kernel<<<dim3(RR_ / 16, 2), 32, 0, stream>>>(h2_16, RR_, 64, 64, qw16, q_b,
                                                        nullptr, Qbuf, DA_);
  gemm_xwT_kernel<<<dim3(RR_ / 16, 2), 32, 0, stream>>>(h2_16, RR_, 64, 64, kw16, k_b,
                                                        nullptr, Kbuf, DA_);
  attn_kernel<<<dim3(65, B_), 256, 0, stream>>>(Qbuf, Kbuf, attp);
  att_reduce_kernel<<<(B_ * TL_ + 255) / 256, 256, 0, stream>>>(attp, att);
  heads_kernel<<<B_, 64, 0, stream>>>(pooled, det_w1, det_b1, det_w2, det_b2,
                                      pred_w1, pred_b1, pred_w2, pred_b2, out);
}
